// OrbitalGenerator_21895743275111
// MI455X (gfx1250) — compile-verified, matrix operand order checked by reading, not run on hardware
//
#include <hip/hip_runtime.h>
#include <hip/hip_bf16.h>

// ---- model dimensions ----
#define O_TOT   1024
#define NNUC    128
#define FDIM    256
#define LAYERS  4
#define HEADS   8
#define DHEAD   32
#define EDIM    7
#define ROWS    (O_TOT * NNUC)        // 131072 (o,n) rows

typedef __bf16 bf16;
typedef __attribute__((ext_vector_type(16))) __bf16 v16bf;
typedef __attribute__((ext_vector_type(8)))  __bf16 v8bf;
typedef __attribute__((ext_vector_type(8)))  float  v8f;
typedef __attribute__((ext_vector_type(4)))  unsigned int u32x4;
typedef __attribute__((ext_vector_type(8)))  int i32x8;
typedef __attribute__((ext_vector_type(4)))  int i32x4;

// Load a 16-element bf16 A/B fragment for v_wmma_f32_16x16x32_bf16.
// Lane convention: lanes<16 hold k-locals {0..7,16..23}, lanes>=16 hold
// {8..15,24..31}; caller passes p = row_base + ktile*32 + (lane>=16 ? 8 : 0).
__device__ __forceinline__ v16bf load_frag16(const bf16* p) {
  v8bf lo = *(const v8bf*)(p);
  v8bf hi = *(const v8bf*)(p + 16);
  v16bf r;
#pragma unroll
  for (int q = 0; q < 8; ++q) { r[q] = lo[q]; r[q + 8] = hi[q]; }
  return r;
}

// ------------------------------------------------------------------
// Tensor Data Mover: 2D tile (rows of tile_d0 8-byte units, tile_d1 rows,
// source row stride stride_d0 8-byte units) -> LDS at lds_addr.
// D# layout per CDNA5 ISA.
// ------------------------------------------------------------------
#if __has_builtin(__builtin_amdgcn_tensor_load_to_lds)
#define HAVE_TDM 1
__device__ __forceinline__ void tdm_load_2d(unsigned lds_addr, const void* gptr,
                                            unsigned tile_d0, unsigned tile_d1,
                                            unsigned stride_d0) {
  unsigned long long ga = (unsigned long long)(size_t)gptr;
  u32x4 g0;
  g0[0] = 1u;                                            // count=1, user mode
  g0[1] = lds_addr;                                      // LDS byte address
  g0[2] = (unsigned)ga;                                  // global_addr[31:0]
  g0[3] = (unsigned)((ga >> 32) & 0x01FFFFFFu) | (2u << 30); // ga[56:32] | type=2
  i32x8 g1;
  g1[0] = (int)(3u << 16);                               // wg_mask=0, data_size=3 (8B)
  g1[1] = (int)((stride_d0 & 0xFFFFu) << 16);            // tensor_dim0[15:0]
  g1[2] = (int)(((stride_d0 >> 16) & 0xFFFFu) |
                ((tile_d1 & 0xFFFFu) << 16));            // tensor_dim0[31:16] | tensor_dim1[15:0]
  g1[3] = (int)((tile_d0 & 0xFFFFu) << 16);              // tensor_dim1[31:16]=0 | tile_dim0
  g1[4] = (int)(tile_d1 & 0xFFFFu);                      // tile_dim1 | tile_dim2=0
  g1[5] = (int)stride_d0;                                // tensor_dim0_stride[31:0]
  g1[6] = 0;                                             // stride[47:32] | dim1_stride lo
  g1[7] = 0;
  i32x4 z4 = {0, 0, 0, 0};
#if defined(__clang_major__) && (__clang_major__ >= 23)
  i32x8 z8 = {0, 0, 0, 0, 0, 0, 0, 0};
  __builtin_amdgcn_tensor_load_to_lds(g0, g1, z4, z4, z8, 0);
#else
  __builtin_amdgcn_tensor_load_to_lds(g0, g1, z4, z4, 0);
#endif
}
#else
#define HAVE_TDM 0
#endif

// ------------------------------------------------------------------
// Prep: per output-orbital o (after permutation) find (nucleus, orbital-type)
// ------------------------------------------------------------------
__global__ __launch_bounds__(1024) void k_prep(const int* __restrict__ charges,
                                               const int* __restrict__ n_up_p,
                                               const int* __restrict__ n_down_p,
                                               int* __restrict__ orbinfo) {
  __shared__ int off_s[NNUC + 1];
  __shared__ int c_s[NNUC];
  int tid = threadIdx.x;
  if (tid < NNUC) c_s[tid] = charges[tid];
  __syncthreads();
  if (tid == 0) {
    int acc = 0;
    for (int n = 0; n < NNUC; ++n) { off_s[n] = acc; acc += c_s[n]; }
    off_s[NNUC] = acc;
  }
  __syncthreads();
  int n_up = n_up_p[0], n_down = n_down_p[0];
  for (int o = tid; o < O_TOT; o += blockDim.x) {
    int e;
    if (o < n_up)                e = o;
    else if (o < 512)            e = o + n_down;
    else if (o < 512 + n_down)   e = o - 512 + n_up;
    else                         e = o;
    int n_found = -1, t_found = -1;
    if (e < off_s[NNUC]) {
      for (int n = 0; n < NNUC; ++n) {
        if (e >= off_s[n] && e < off_s[n] + c_s[n]) {
          n_found = n;
          int i = e - off_s[n];
          t_found = c_s[n] * (c_s[n] - 1) / 2 + i;
          break;
        }
      }
    }
    orbinfo[2 * o]     = n_found;
    orbinfo[2 * o + 1] = t_found;
  }
}

// ------------------------------------------------------------------
// Pack W [L][256][256] f32 -> WMMA-B lane layout bf16: [L][kt(8)][ct(16)][lane(32)][16]
// ------------------------------------------------------------------
__global__ __launch_bounds__(256) void k_pack_w(const float* __restrict__ W,
                                                bf16* __restrict__ Wp) {
  int t = blockIdx.x * 256 + threadIdx.x;       // 0 .. 262143
  int p    = t & 15;
  int lane = (t >> 4) & 31;
  int ct   = (t >> 9) & 15;
  int kt   = (t >> 13) & 7;
  int l    = t >> 16;
  int base = (lane >= 16) ? 8 : 0;
  int klocal = (p < 8) ? (base + p) : (base + 16 + (p - 8));
  int k = kt * 32 + klocal;
  int col = ct * 16 + (lane & 15);
  Wp[t] = (bf16)W[((size_t)l * FDIM + k) * FDIM + col];
}

// ------------------------------------------------------------------
// Edge features + per-layer/head edge weights (bf16, [L][H][i][j])
// ------------------------------------------------------------------
__global__ __launch_bounds__(128) void k_edge_ew(const float* __restrict__ coords,
                                                 const unsigned char* __restrict__ nm,
                                                 const float* __restrict__ W_edge,
                                                 bf16* __restrict__ ewb) {
  __shared__ float we[LAYERS * EDIM * HEADS];   // 224
  int i = blockIdx.x, j = threadIdx.x;
  if (j < LAYERS * EDIM * HEADS) we[j] = W_edge[j];
  __syncthreads();
  float dx = coords[i * 3 + 0] - coords[j * 3 + 0];
  float dy = coords[i * 3 + 1] - coords[j * 3 + 1];
  float dz = coords[i * 3 + 2] - coords[j * 3 + 2];
  float nrm = sqrtf(dx * dx + dy * dy + dz * dz + 1e-12f);
  float inv = 1.0f / (1.0f + nrm);
  float e[EDIM];
  e[0] = dx * inv; e[1] = dy * inv; e[2] = dz * inv;
  e[3] = log1pf(nrm);
  e[4] = 1.0f / (1.0f + __expf(-(nrm - 2.0f)));
  e[5] = 1.0f / (1.0f + __expf(-(nrm - 4.0f)));
  e[6] = 1.0f / (1.0f + __expf(-(nrm - 6.0f)));
  float msk = (nm[i] && nm[j]) ? 1.0f : 0.0f;
#pragma unroll
  for (int l = 0; l < LAYERS; ++l)
#pragma unroll
    for (int h = 0; h < HEADS; ++h) {
      float s = 0.0f;
#pragma unroll
      for (int k = 0; k < EDIM; ++k) s += e[k] * we[(l * EDIM + k) * HEADS + h];
      ewb[(((size_t)l * HEADS + h) * NNUC + i) * NNUC + j] = (bf16)(s * msk);
    }
}

// ------------------------------------------------------------------
// nucterm[l][i][f] = nuc_feats[i,:] @ W_nuc[l]  (tiny, VALU)
// ------------------------------------------------------------------
__global__ __launch_bounds__(256) void k_nucterm(const float* __restrict__ nuc_feats,
                                                 const float* __restrict__ W_nuc,
                                                 float* __restrict__ nucterm) {
  int b = blockIdx.x;             // l*128 + i
  int l = b >> 7, i = b & 127;
  int f = threadIdx.x;
  float acc = 0.0f;
  const float* nf = nuc_feats + (size_t)i * FDIM;
  const float* w  = W_nuc + (size_t)l * FDIM * FDIM + f;
  for (int k = 0; k < FDIM; ++k) acc += nf[k] * w[(size_t)k * FDIM];
  nucterm[(size_t)b * FDIM + f] = acc;
}

// ------------------------------------------------------------------
// feats[o,n,f] = (n == nuc(o)) ? W_orb[type(o),f] * mask : 0
// ------------------------------------------------------------------
__global__ __launch_bounds__(256) void k_init_feats(const int* __restrict__ orbinfo,
                                                    const float* __restrict__ W_orb,
                                                    const unsigned char* __restrict__ om,
                                                    const unsigned char* __restrict__ nm,
                                                    float* __restrict__ feats) {
  int o = blockIdx.x, f = threadIdx.x;
  int n_o = orbinfo[2 * o], t = orbinfo[2 * o + 1];
  float w = 0.0f;
  if (n_o >= 0 && om[o]) w = W_orb[(size_t)t * FDIM + f];
  for (int n = 0; n < NNUC; ++n) {
    float v = (n == n_o && nm[n]) ? w : 0.0f;
    feats[((size_t)o * NNUC + n) * FDIM + f] = v;
  }
}

// ------------------------------------------------------------------
// OrbLayerNorm stats: mean per (o,n); rsd[o] = rsqrt(max_n var + 1)
// ------------------------------------------------------------------
__global__ __launch_bounds__(1024) void k_ln_stats(const float* __restrict__ x,
                                                   float* __restrict__ mean,
                                                   float* __restrict__ rsd) {
  int o = blockIdx.x;
  int lane = threadIdx.x & 31, wv = threadIdx.x >> 5;   // 32 waves
  __shared__ float vsh[NNUC];
#pragma unroll
  for (int rr = 0; rr < 4; ++rr) {
    int n = wv * 4 + rr;
    const float* row = x + ((size_t)o * NNUC + n) * FDIM;
    float s = 0.0f, sq = 0.0f;
#pragma unroll
    for (int q = 0; q < 8; ++q) {
      float v = row[q * 32 + lane];
      s += v; sq += v * v;
    }
#pragma unroll
    for (int off = 16; off > 0; off >>= 1) {
      s  += __shfl_xor(s,  off, 32);
      sq += __shfl_xor(sq, off, 32);
    }
    if (lane == 0) {
      float mu = s * (1.0f / FDIM);
      mean[(size_t)o * NNUC + n] = mu;
      vsh[n] = sq * (1.0f / FDIM) - mu * mu;
    }
  }
  __syncthreads();
  if (threadIdx.x == 0) {
    float mx = vsh[0];
    for (int n = 1; n < NNUC; ++n) mx = fmaxf(mx, vsh[n]);
    rsd[o] = rsqrtf(mx + 1.0f);
  }
}

// ------------------------------------------------------------------
// Normalize (optionally tanh) -> bf16 rows
// ------------------------------------------------------------------
template <bool TANH>
__global__ __launch_bounds__(256) void k_norm(const float* __restrict__ x,
                                              const float* __restrict__ mean,
                                              const float* __restrict__ rsd,
                                              bf16* __restrict__ out) {
  size_t r = blockIdx.x;
  float mu = mean[r], sc = rsd[r >> 7];
  float v = (x[r * FDIM + threadIdx.x] - mu) * sc;
  if (TANH) v = tanhf(v);
  out[r * FDIM + threadIdx.x] = (bf16)v;
}

// ------------------------------------------------------------------
// GEMM over rows: C[r,g] = A[r,:] @ Wp[:,g], r in [0,131072), K=256.
// The block's 64-col B panel (32 KB) is shared by all 8 waves -> TDM-stage it
// into LDS once, then WMMA with A double-buffered from global.
// MODE 0: write vals into valsT[h][o][d][n] (bf16, j-fastest for GEMM2 B).
// MODE 1: feats[r,g] += C[r,g] * mask(o,n).
// Grid = (1024, 4), block = 8 waves; wave owns 16 rows x 64 cols.
// ------------------------------------------------------------------
template <int MODE>
__global__ __launch_bounds__(256) void k_gemm(const bf16* __restrict__ A,
                                              const bf16* __restrict__ Wp,
                                              float* __restrict__ feats,
                                              bf16* __restrict__ valsT,
                                              const unsigned char* __restrict__ om,
                                              const unsigned char* __restrict__ nm) {
  __shared__ __align__(32) bf16 sB[16384];      // [kt(8)][j(4)][lane(32)][16]
  const int lane = threadIdx.x & 31;
  const int wv   = threadIdx.x >> 5;
  const int cg   = blockIdx.y;                  // 64-col group
#if HAVE_TDM
  if (wv == 0) {
    // 8 rows (one per kt) of 4 KB, source row stride 16 KB.
    tdm_load_2d((unsigned)(size_t)&sB[0], Wp + (size_t)cg * 2048,
                /*tile_d0=*/512, /*tile_d1=*/8, /*stride_d0=*/2048);
    __builtin_amdgcn_s_wait_tensorcnt(0);
  }
#else
  for (int t = threadIdx.x; t < 16384; t += 256) {
    int kt = t >> 11, rem = t & 2047;
    sB[t] = Wp[(size_t)(kt * 16 + cg * 4) * 512 + rem];
  }
#endif
  __syncthreads();

  const int r0 = (blockIdx.x * 8 + wv) * 16;
  const int m  = lane & 15;
  const int kg = (lane >> 4) * 8;               // k-group offset (0 or 8)
  v8f acc[4] = {};
  const bf16* arow = A + (size_t)(r0 + m) * FDIM + kg;
  v16bf a_cur = load_frag16(arow);
#pragma unroll
  for (int kt = 0; kt < 8; ++kt) {
    v16bf a_nxt = a_cur;
    if (kt + 1 < 8) a_nxt = load_frag16(arow + (kt + 1) * 32);  // overlap with WMMAs
    const bf16* bb = &sB[kt * 2048 + lane * 16];
#pragma unroll
    for (int j = 0; j < 4; ++j) {
      v16bf b = *(const v16bf*)(bb + j * 512);
      acc[j] = __builtin_amdgcn_wmma_f32_16x16x32_bf16(
          false, a_cur, false, b, (short)0, acc[j], false, false);
    }
    a_cur = a_nxt;
  }

  const int o  = r0 >> 7;
  const int nb = (r0 & 127) + kg;               // 8 consecutive n per lane
#pragma unroll
  for (int j = 0; j < 4; ++j) {
    int g = (cg * 4 + j) * 16 + m;
    if (MODE == 0) {
      int h = g >> 5, d = g & 31;
      v8bf out;
#pragma unroll
      for (int v = 0; v < 8; ++v) out[v] = (bf16)acc[j][v];
      size_t idx = (((size_t)h * O_TOT + o) * DHEAD + d) * NNUC + nb;
      *(v8bf*)(valsT + idx) = out;
    } else {
      float omv = om[o] ? 1.0f : 0.0f;
#pragma unroll
      for (int v = 0; v < 8; ++v) {
        int n = nb + v;
        float mval = nm[n] ? omv : 0.0f;
        size_t idx = ((size_t)o * NNUC + n) * FDIM + g;
        feats[idx] += acc[j][v] * mval;
      }
    }
  }
}

// ------------------------------------------------------------------
// msg[o,i,h*32+d] = sum_j ew[h,i,j] * valsT[h][c][j] + nucterm[i,f], c=o*32+d.
// Per head: M=128(i), K=128(j), N=32768. The block's 64-col B panel (16 KB,
// contiguous in valsT) is shared by all 8 waves -> TDM-stage into LDS.
// Grid = (8 heads, 512 col groups), block = 8 waves (all 8 i-tiles).
// ------------------------------------------------------------------
__global__ __launch_bounds__(256) void k_gemm_msg(const bf16* __restrict__ ewl,
                                                  const bf16* __restrict__ valsT,
                                                  const float* __restrict__ nucterm_l,
                                                  float* __restrict__ msg) {
  __shared__ __align__(32) bf16 sB[8192];       // [c_local(64)][j(128)]
  const int h    = blockIdx.x;
  const int cg   = blockIdx.y;                  // 64-col group
  const int lane = threadIdx.x & 31;
  const int wv   = threadIdx.x >> 5;
  const bf16* bbase = valsT + (size_t)h * O_TOT * DHEAD * NNUC;
#if HAVE_TDM
  if (wv == 0) {
    tdm_load_2d((unsigned)(size_t)&sB[0], bbase + (size_t)cg * 8192,
                /*tile_d0=*/2048, /*tile_d1=*/1, /*stride_d0=*/2048);
    __builtin_amdgcn_s_wait_tensorcnt(0);
  }
#else
  for (int t = threadIdx.x; t < 8192; t += 256) sB[t] = bbase[(size_t)cg * 8192 + t];
#endif
  __syncthreads();

  const int i0 = wv * 16;
  const int m  = lane & 15;
  const int kg = (lane >> 4) * 8;
  v8f acc[4] = {};
  const bf16* arow = ewl + ((size_t)(h * NNUC + i0 + m)) * NNUC + kg;
  v16bf a_cur = load_frag16(arow);
#pragma unroll
  for (int kt = 0; kt < 4; ++kt) {
    v16bf a_nxt = a_cur;
    if (kt + 1 < 4) a_nxt = load_frag16(arow + (kt + 1) * 32);
#pragma unroll
    for (int j = 0; j < 4; ++j) {
      const bf16* bp = &sB[(j * 16 + m) * NNUC + kt * 32 + kg];
      v16bf b = load_frag16(bp);
      acc[j] = __builtin_amdgcn_wmma_f32_16x16x32_bf16(
          false, a_cur, false, b, (short)0, acc[j], false, false);
    }
    a_cur = a_nxt;
  }
#pragma unroll
  for (int j = 0; j < 4; ++j) {
    int c = cg * 64 + j * 16 + m;
    int o = c >> 5, d = c & 31;
    int f = h * DHEAD + d;
#pragma unroll
    for (int v = 0; v < 8; ++v) {
      int i = i0 + v + kg;
      size_t idx = ((size_t)o * NNUC + i) * FDIM + f;
      msg[idx] = acc[j][v] + nucterm_l[(size_t)i * FDIM + f];
    }
  }
}

// ------------------------------------------------------------------
extern "C" void kernel_launch(void* const* d_in, const int* in_sizes, int n_in,
                              void* d_out, int out_size, void* d_ws, size_t ws_size,
                              hipStream_t stream) {
  const float*         coords    = (const float*)d_in[0];
  const int*           charges   = (const int*)d_in[1];
  const unsigned char* nuc_mask  = (const unsigned char*)d_in[2];
  const int*           n_up      = (const int*)d_in[3];
  const int*           n_down    = (const int*)d_in[4];
  const float*         nuc_feats = (const float*)d_in[5];
  const unsigned char* orb_mask  = (const unsigned char*)d_in[6];
  const float*         W_orb     = (const float*)d_in[7];
  const float*         W_edge    = (const float*)d_in[8];
  const float*         W_val     = (const float*)d_in[9];
  const float*         W_nuc     = (const float*)d_in[10];
  const float*         W_out     = (const float*)d_in[11];
  float* feats = (float*)d_out;                 // [O, N, F] f32, 134 MB

  // ---- workspace partition ----
  char* ws = (char*)d_ws;
  const size_t SZ_H    = (size_t)ROWS * FDIM * sizeof(bf16);           // 67 MB
  const size_t SZ_VT   = (size_t)HEADS * O_TOT * DHEAD * NNUC * sizeof(bf16); // 67 MB
  const size_t SZ_MSG  = (size_t)ROWS * FDIM * sizeof(float);          // 134 MB
  const size_t SZ_MEAN = (size_t)ROWS * sizeof(float);
  const size_t SZ_RSD  = (size_t)O_TOT * sizeof(float);
  const size_t SZ_WP   = (size_t)LAYERS * FDIM * FDIM * sizeof(bf16);
  const size_t SZ_EW   = (size_t)LAYERS * HEADS * NNUC * NNUC * sizeof(bf16);
  const size_t SZ_NT   = (size_t)LAYERS * NNUC * FDIM * sizeof(float);
  bf16*  h_bf   = (bf16*)(ws);                      ws += SZ_H;
  bf16*  valsT  = (bf16*)(ws);                      ws += SZ_VT;
  float* msg    = (float*)(ws);                     ws += SZ_MSG;
  float* mean   = (float*)(ws);                     ws += SZ_MEAN;
  float* rsd    = (float*)(ws);                     ws += SZ_RSD;
  bf16*  Wpv    = (bf16*)(ws);                      ws += SZ_WP;
  bf16*  Wpo    = (bf16*)(ws);                      ws += SZ_WP;
  bf16*  ewb    = (bf16*)(ws);                      ws += SZ_EW;
  float* nuct   = (float*)(ws);                     ws += SZ_NT;
  int*   orbinf = (int*)(ws);

  // ---- setup (cheap, once per call) ----
  k_prep<<<1, 1024, 0, stream>>>(charges, n_up, n_down, orbinf);
  k_pack_w<<<1024, 256, 0, stream>>>(W_val, Wpv);
  k_pack_w<<<1024, 256, 0, stream>>>(W_out, Wpo);
  k_edge_ew<<<NNUC, NNUC, 0, stream>>>(coords, nuc_mask, W_edge, ewb);
  k_nucterm<<<LAYERS * NNUC, 256, 0, stream>>>(nuc_feats, W_nuc, nuct);
  k_init_feats<<<O_TOT, 256, 0, stream>>>(orbinf, W_orb, orb_mask, nuc_mask, feats);

  // ---- layers ----
  for (int l = 0; l < LAYERS; ++l) {
    k_ln_stats<<<O_TOT, 1024, 0, stream>>>(feats, mean, rsd);
    k_norm<false><<<ROWS, 256, 0, stream>>>(feats, mean, rsd, h_bf);
    k_gemm<0><<<dim3(1024, 4), 256, 0, stream>>>(
        h_bf, Wpv + (size_t)l * FDIM * FDIM, feats, valsT, orb_mask, nuc_mask);
    k_gemm_msg<<<dim3(HEADS, 512), 256, 0, stream>>>(
        ewb + (size_t)l * HEADS * NNUC * NNUC, valsT,
        nuct + (size_t)l * NNUC * FDIM, msg);
    k_ln_stats<<<O_TOT, 1024, 0, stream>>>(msg, mean, rsd);
    k_norm<true><<<ROWS, 256, 0, stream>>>(msg, mean, rsd, h_bf);
    k_gemm<1><<<dim3(1024, 4), 256, 0, stream>>>(
        h_bf, Wpo + (size_t)l * FDIM * FDIM, feats, valsT, orb_mask, nuc_mask);
  }
}
